// TGN_23356032155945
// MI455X (gfx1250) — matrix operand order repaired and checked by reading, not compiled, not run on hardware
//
#include <hip/hip_runtime.h>
#include <hip/hip_fp16.h>
#include <math.h>

// ---------------------------------------------------------------- types
typedef __attribute__((ext_vector_type(16))) _Float16 v16h;
typedef __attribute__((ext_vector_type(8)))  float    v8f;

#define D_    128
#define T_    128
#define E_    128
#define MSG_  512      // 2D + E + T
#define QD_   256      // D + T
#define KD_   384      // D + E + T
#define TD3_  384      // 3*D
#define KNB_  20
#define HEADS_ 2
#define HD_   128      // QD_/HEADS
#define SB_   8        // seeds per attention block
#define ROWS_ (SB_ * KNB_)   // 160 k_in rows per block = 10 row-tiles

// ---------------------------------------------------------------- WMMA helpers
// A tile (LDS, row-major, 16 rows x lda halfs). 16-bit A 16x32 layout:
// lane: m = lane&15, g = lane>>4 ; element j: j<8 -> K=g*8+j ; j>=8 -> K=16+g*8+(j-8)
__device__ inline v16h frag_a(const _Float16* A, int lda, int k0, int lane) {
    const _Float16* r = A + (lane & 15) * lda + k0 + (lane >> 4) * 8;
    v16h v;
#pragma unroll
    for (int j = 0; j < 8; ++j) { v[j] = r[j]; v[8 + j] = r[16 + j]; }
    return v;
}

// B from column-major weights: Bt[col][K], ldb = K extent.
// 16-bit B 32x16 layout: lane: n=lane&15, g=lane>>4 ; element j: j<8 -> K=g*8+j ; j>=8 -> K=16+g*8+(j-8)
__device__ inline v16h frag_b(const _Float16* Bt, int ldb, int col0, int k0, int lane) {
    const _Float16* p = Bt + (size_t)(col0 + (lane & 15)) * ldb + k0 + (lane >> 4) * 8;
    v16h v;
#pragma unroll
    for (int j = 0; j < 8; ++j) { v[j] = p[j]; v[8 + j] = p[16 + j]; }
    return v;
}

__device__ inline v8f wmma_acc(v8f c, v16h a, v16h b) {
    return __builtin_amdgcn_wmma_f32_16x16x32_f16(false, a, false, b, (short)0, c,
                                                  false, false);
}

__device__ inline float sigmoidf_(float x) { return 1.f / (1.f + __expf(-x)); }

// ---------------------------------------------------------------- small kernels
__global__ void k_zero_map(int* map, int n) {
    int i = blockIdx.x * blockDim.x + threadIdx.x;
    if (i < n) map[i] = 0;
}

__global__ void k_scatter_map(const int* unids, int* map, int U) {
    int i = blockIdx.x * blockDim.x + threadIdx.x;
    if (i < U) map[unids[i]] = i + 1;
}

// f16 copies of all GEMM weights, stored column-major (B-col contiguous in K).
// W_ih / W_hh rows are already B-columns (x @ W.T); Wk/Wv/Wq/Wo/W1/W2 transposed.
__global__ void k_prep(const float* Wih, const float* Whh, const float* Wk,
                       const float* Wv, const float* Wq, const float* Wo,
                       const float* W1, const float* W2,
                       _Float16* wih_h, _Float16* whh_h, _Float16* wkT,
                       _Float16* wvT, _Float16* wqT, _Float16* woT,
                       _Float16* w1T, _Float16* w2T) {
    int stride = gridDim.x * blockDim.x;
    int i = blockIdx.x * blockDim.x + threadIdx.x;
    for (int t = i; t < TD3_ * MSG_; t += stride) wih_h[t] = (_Float16)Wih[t];
    for (int t = i; t < TD3_ * D_; t += stride)  whh_h[t] = (_Float16)Whh[t];
    for (int t = i; t < QD_ * KD_; t += stride) {
        int n = t / KD_, k = t % KD_;
        wkT[t] = (_Float16)Wk[(size_t)k * QD_ + n];
        wvT[t] = (_Float16)Wv[(size_t)k * QD_ + n];
    }
    for (int t = i; t < QD_ * QD_; t += stride) {
        int n = t / QD_, k = t % QD_;
        wqT[t] = (_Float16)Wq[(size_t)k * QD_ + n];
        woT[t] = (_Float16)Wo[(size_t)k * QD_ + n];
    }
    for (int t = i; t < D_ * KD_; t += stride) {
        int n = t / KD_, k = t % KD_;
        w1T[t] = (_Float16)W1[(size_t)k * D_ + n];
    }
    for (int t = i; t < D_ * D_; t += stride) {
        int n = t / D_, k = t % D_;
        w2T[t] = (_Float16)W2[(size_t)k * D_ + n];
    }
}

// ---------------------------------------------------------------- GRU memory update
// one block = 16 rows of U ; 256 threads = 8 waves ; 3 col-tiles per wave
__global__ void __launch_bounds__(256) k_gru(
    const float* memory, const int* unids, const float* umsg,
    const _Float16* wih_h, const _Float16* whh_h,
    const float* b_ih, const float* b_hh, float* mem_upd, int U) {
    extern __shared__ char smem[];
    _Float16* Amsg = (_Float16*)smem;              // 16*512 f16
    _Float16* Ah   = Amsg + 16 * MSG_;             // 16*128 f16
    float* hf      = (float*)(Ah + 16 * D_);       // 16*128 f32
    float* acc_i   = hf + 16 * D_;                 // 16*384 f32
    float* acc_h   = acc_i + 16 * TD3_;            // 16*384 f32

    const int tid = threadIdx.x;
    const int r0 = blockIdx.x * 16;

    for (int e = tid; e < 16 * MSG_; e += 256) {
        int r = e / MSG_, c = e - r * MSG_;
        int row = r0 + r;
        Amsg[e] = (row < U) ? (_Float16)umsg[(size_t)row * MSG_ + c] : (_Float16)0.f;
    }
    for (int e = tid; e < 16 * D_; e += 256) {
        int r = e / D_, c = e - r * D_;
        int row = r0 + r;
        float hv = 0.f;
        if (row < U) hv = memory[(size_t)unids[row] * D_ + c];
        hf[e] = hv;
        Ah[e] = (_Float16)hv;
    }
    __syncthreads();

    const int wave = tid >> 5, lane = tid & 31;
    for (int ct = wave; ct < TD3_ / 16; ct += 8) {   // 24 col tiles / 8 waves
        v8f ci = {};
        v8f ch = {};
#pragma unroll 4
        for (int k0 = 0; k0 < MSG_; k0 += 32)
            ci = wmma_acc(ci, frag_a(Amsg, MSG_, k0, lane),
                          frag_b(wih_h, MSG_, ct * 16, k0, lane));
#pragma unroll
        for (int k0 = 0; k0 < D_; k0 += 32)
            ch = wmma_acc(ch, frag_a(Ah, D_, k0, lane),
                          frag_b(whh_h, D_, ct * 16, k0, lane));
        const int n = ct * 16 + (lane & 15);
        const int g = lane >> 4;
#pragma unroll
        for (int r = 0; r < 8; ++r) {
            int m = r + 8 * g;
            acc_i[m * TD3_ + n] = ci[r];
            acc_h[m * TD3_ + n] = ch[r];
        }
    }
    __syncthreads();

    for (int e = tid; e < 16 * D_; e += 256) {
        int r = e / D_, d = e - r * D_;
        int row = r0 + r;
        if (row >= U) continue;
        float ir = acc_i[r * TD3_ + d]          + b_ih[d];
        float iz = acc_i[r * TD3_ + D_ + d]     + b_ih[D_ + d];
        float ig = acc_i[r * TD3_ + 2 * D_ + d] + b_ih[2 * D_ + d];
        float hr = acc_h[r * TD3_ + d]          + b_hh[d];
        float hz = acc_h[r * TD3_ + D_ + d]     + b_hh[D_ + d];
        float hg = acc_h[r * TD3_ + 2 * D_ + d] + b_hh[2 * D_ + d];
        float rg = sigmoidf_(ir + hr);
        float z  = sigmoidf_(iz + hz);
        float gg = tanhf(ig + rg * hg);
        mem_upd[(size_t)row * D_ + d] = (1.f - z) * gg + z * hf[e];
    }
}

// ---------------------------------------------------------------- fused attention+merge
// one block (256 threads / 8 waves) per 8 seeds; all GEMMs via WMMA with
// register-resident B fragments (reused across row-tiles).
__global__ void __launch_bounds__(256) k_attn(
    const float* staticf, const float* memory, const int* map, const float* mem_upd,
    const int* seeds, const float* stimes, const int* nnids, const float* ntimes,
    const float* nfeats, const unsigned char* nmask,
    const float* w_t, const float* b_t,
    const _Float16* wkT, const _Float16* wvT, const _Float16* wqT,
    const _Float16* woT, const _Float16* w1T, const _Float16* w2T,
    const float* b1, const float* b2, float* out) {
    extern __shared__ char smem[];
    _Float16* kin  = (_Float16*)smem;                 // 160*384 f16
    _Float16* kv16 = kin + ROWS_ * KD_;               // 160*512 f16
    _Float16* qin  = kv16 + ROWS_ * 512;              // 16*256 f16 (rows>=8 zero)
    float* qv   = (float*)(qin + 16 * QD_);           // 8*256 f32
    float* nf   = qv + SB_ * QD_;                     // 8*128 f32
    float* sc   = nf + SB_ * D_;                      // 8*2*20
    float* sw   = sc + SB_ * HEADS_ * KNB_;           // 8*2*20
    int*  flags = (int*)(sw + SB_ * HEADS_ * KNB_);   // 8
    unsigned char* mb = (unsigned char*)(flags + SB_);// 160
    // aliased into kin region after the K/V stage:
    _Float16* af16 = kin;                 // 16*256  attention rows (rows>=8 zero)
    _Float16* catf = af16 + 16 * QD_;     // 16*384  [wo_out | node_feat]
    _Float16* h1s  = catf + 16 * KD_;     // 16*128

    const int tid = threadIdx.x;
    const int s0 = blockIdx.x * SB_;
    const int wave = tid >> 5, lane = tid & 31;

    if (tid < SB_) flags[tid] = 0;
    __syncthreads();

    // ---- stage 0: build node_feat, q_in, k_in tiles
    for (int e = tid; e < SB_ * D_; e += 256) {     // node_feat + q_in left half
        int i = e / D_, d = e - i * D_;
        int node = seeds[s0 + i];
        int mi = map[node];
        const float* msrc = (mi > 0) ? (mem_upd + (size_t)(mi - 1) * D_)
                                     : (memory + (size_t)node * D_);
        float v = staticf[(size_t)node * D_ + d] + msrc[d];
        nf[e] = v;
        qin[i * QD_ + d] = (_Float16)v;
        qin[i * QD_ + D_ + d] = (_Float16)__cosf(b_t[d]);   // Time2Vec(0)
    }
    for (int e = tid; e < 8 * QD_; e += 256)        // zero q_in rows 8..15
        qin[8 * QD_ + e] = (_Float16)0.f;
    for (int e = tid; e < ROWS_; e += 256)          // masks + any-mask flags
        { unsigned char m = nmask[(size_t)s0 * KNB_ + e];
          mb[e] = m; if (m) atomicOr(&flags[e / KNB_], 1); }
    for (int e = tid; e < ROWS_ * KD_; e += 256) {  // k_in rows (8 seeds x 20 nbrs)
        int r = e / KD_, c = e - r * KD_;
        size_t gk = (size_t)s0 * KNB_ + r;          // global (seed,nbr) index
        float v;
        if (c < D_) {
            v = staticf[(size_t)nnids[gk] * D_ + c];
        } else if (c < D_ + E_) {
            v = nfeats[gk * E_ + (c - D_)];
        } else {
            int d = c - D_ - E_;
            float dt = nmask[gk] ? (stimes[s0 + r / KNB_] - ntimes[gk]) : 0.f;
            v = __cosf(dt * w_t[d] + b_t[d]);
        }
        kin[e] = (_Float16)v;
    }
    __syncthreads();

    // ---- stage 1: q = q_in @ Wq  (1 row-tile x 16 col-tiles)
    for (int ci = 0; ci < 2; ++ci) {
        const int ct = wave + ci * 8;
        v8f c = {};
#pragma unroll
        for (int k0 = 0; k0 < QD_; k0 += 32)
            c = wmma_acc(c, frag_a(qin, QD_, k0, lane),
                         frag_b(wqT, QD_, ct * 16, k0, lane));
        const int n = ct * 16 + (lane & 15);
        if ((lane >> 4) == 0) {
#pragma unroll
            for (int r = 0; r < 8; ++r) qv[r * QD_ + n] = c[r];
        }
    }

    // ---- stage 2: K|V = k_in @ [Wk|Wv] ; B fragments register-resident,
    // reused across all 10 row-tiles.
    for (int ci = 0; ci < 4; ++ci) {
        const int ct = wave + ci * 8;               // 0..31 over 512 cols
        const int col = ct * 16;
        const _Float16* Bt = (col < QD_) ? wkT : wvT;
        const int bcol = (col < QD_) ? col : col - QD_;
        v16h Bf[12];
#pragma unroll
        for (int kk = 0; kk < 12; ++kk)
            Bf[kk] = frag_b(Bt, KD_, bcol, kk * 32, lane);
        for (int rt = 0; rt < ROWS_ / 16; ++rt) {
            const _Float16* At = kin + rt * 16 * KD_;
            v8f c = {};
#pragma unroll
            for (int kk = 0; kk < 12; ++kk)
                c = wmma_acc(c, frag_a(At, KD_, kk * 32, lane), Bf[kk]);
            const int n = col + (lane & 15);
            const int g = lane >> 4;
#pragma unroll
            for (int r = 0; r < 8; ++r)
                kv16[(size_t)(rt * 16 + r + 8 * g) * 512 + n] = (_Float16)c[r];
        }
    }
    __syncthreads();

    // ---- stage 3: masked scores, softmax, weighted sum -> af16
    for (int e = tid; e < SB_ * HEADS_ * KNB_; e += 256) {
        int i = e / (HEADS_ * KNB_);
        int h = (e / KNB_) % HEADS_;
        int k = e % KNB_;
        float acc = 0.f;
        const _Float16* kr = kv16 + (size_t)(i * KNB_ + k) * 512 + h * HD_;
        for (int d = 0; d < HD_; ++d) acc += qv[i * QD_ + h * HD_ + d] * (float)kr[d];
        sc[e] = mb[i * KNB_ + k] ? acc * 0.08838834764831845f : -1e30f;
    }
    __syncthreads();
    if (tid < SB_ * HEADS_) {                        // softmax over K
        float* p = sc + tid * KNB_;
        float* q = sw + tid * KNB_;
        float mx = -3.4e38f;
        for (int k = 0; k < KNB_; ++k) mx = fmaxf(mx, p[k]);
        float sum = 0.f;
        for (int k = 0; k < KNB_; ++k) { float e = __expf(p[k] - mx); q[k] = e; sum += e; }
        float inv = 1.f / sum;
        for (int k = 0; k < KNB_; ++k) q[k] *= inv;
    }
    __syncthreads();
    for (int e = tid; e < 16 * QD_; e += 256) {      // attn rows (and zero pad rows)
        int i = e / QD_, t = e - i * QD_;
        float acc = 0.f;
        if (i < SB_) {
            int h = t / HD_, d = t - h * HD_;
            for (int k = 0; k < KNB_; ++k)
                acc += sw[(i * HEADS_ + h) * KNB_ + k] *
                       (float)kv16[(size_t)(i * KNB_ + k) * 512 + QD_ + h * HD_ + d];
        }
        af16[e] = (_Float16)acc;                     // aliases dead kin region
    }
    for (int e = tid; e < 16 * KD_; e += 256) {      // catf: nf right half + zero pad
        int i = e / KD_, c = e - i * KD_;
        float v = 0.f;
        if (i < SB_ && c >= QD_) v = nf[i * D_ + (c - QD_)];
        if (c >= QD_ || i >= SB_) catf[e] = (_Float16)v;   // cols<QD_ filled by stage 4
    }
    for (int e = tid; e < 8 * D_; e += 256)          // zero h1 rows 8..15
        h1s[8 * D_ + e] = (_Float16)0.f;
    __syncthreads();

    // ---- stage 4: attn @ Wo -> catf[:, 0:256] (zeroing fully-masked rows)
    for (int ci = 0; ci < 2; ++ci) {
        const int ct = wave + ci * 8;
        v8f c = {};
#pragma unroll
        for (int k0 = 0; k0 < QD_; k0 += 32)
            c = wmma_acc(c, frag_a(af16, QD_, k0, lane),
                         frag_b(woT, QD_, ct * 16, k0, lane));
        const int n = ct * 16 + (lane & 15);
        if ((lane >> 4) == 0) {
#pragma unroll
            for (int r = 0; r < 8; ++r)
                catf[r * KD_ + n] = (_Float16)(flags[r] ? c[r] : 0.f);
        }
    }
    __syncthreads();

    // ---- stage 5: h1 = relu(cat @ W1 + b1)
    {
        const int ct = wave;                         // 8 col tiles of 128
        v8f c = {};
#pragma unroll
        for (int k0 = 0; k0 < KD_; k0 += 32)
            c = wmma_acc(c, frag_a(catf, KD_, k0, lane),
                         frag_b(w1T, KD_, ct * 16, k0, lane));
        const int n = ct * 16 + (lane & 15);
        if ((lane >> 4) == 0) {
#pragma unroll
            for (int r = 0; r < 8; ++r)
                h1s[r * D_ + n] = (_Float16)fmaxf(c[r] + b1[n], 0.f);
        }
    }
    __syncthreads();

    // ---- stage 6: out = h1 @ W2 + b2
    {
        const int ct = wave;
        v8f c = {};
#pragma unroll
        for (int k0 = 0; k0 < D_; k0 += 32)
            c = wmma_acc(c, frag_a(h1s, D_, k0, lane),
                         frag_b(w2T, D_, ct * 16, k0, lane));
        const int n = ct * 16 + (lane & 15);
        if ((lane >> 4) == 0) {
#pragma unroll
            for (int r = 0; r < 8; ++r)
                out[(size_t)(s0 + r) * D_ + n] = c[r] + b2[n];
        }
    }
}

// ---------------------------------------------------------------- launch
extern "C" void kernel_launch(void* const* d_in, const int* in_sizes, int n_in,
                              void* d_out, int out_size, void* d_ws, size_t ws_size,
                              hipStream_t stream) {
    const float* staticf = (const float*)d_in[0];
    const float* memory  = (const float*)d_in[1];
    const int*   unids   = (const int*)d_in[2];
    const float* umsg    = (const float*)d_in[3];
    const int*   seeds   = (const int*)d_in[4];
    const float* stimes  = (const float*)d_in[5];
    const int*   nnids   = (const int*)d_in[6];
    const float* ntimes  = (const float*)d_in[7];
    const float* nfeats  = (const float*)d_in[8];
    const unsigned char* nmask = (const unsigned char*)d_in[9];
    const float* w_t = (const float*)d_in[10];
    const float* b_t = (const float*)d_in[11];
    const float* Wih = (const float*)d_in[12];
    const float* bih = (const float*)d_in[13];
    const float* Whh = (const float*)d_in[14];
    const float* bhh = (const float*)d_in[15];
    const float* Wq  = (const float*)d_in[16];
    const float* Wk  = (const float*)d_in[17];
    const float* Wv  = (const float*)d_in[18];
    const float* Wo  = (const float*)d_in[19];
    const float* W1  = (const float*)d_in[20];
    const float* b1  = (const float*)d_in[21];
    const float* W2  = (const float*)d_in[22];
    const float* b2  = (const float*)d_in[23];

    const int N = in_sizes[0] / D_;
    const int U = in_sizes[2];
    const int S = in_sizes[4];

    // ---- workspace carve-up
    uintptr_t base = (uintptr_t)d_ws;
    auto alignup = [](uintptr_t p) { return (p + 255u) & ~(uintptr_t)255u; };
    int* map = (int*)base;              base = alignup(base + (size_t)N * 4);
    float* mem_upd = (float*)base;      base = alignup(base + (size_t)U * D_ * 4);
    _Float16* wih_h = (_Float16*)base;  base = alignup(base + (size_t)TD3_ * MSG_ * 2);
    _Float16* whh_h = (_Float16*)base;  base = alignup(base + (size_t)TD3_ * D_ * 2);
    _Float16* wkT = (_Float16*)base;    base = alignup(base + (size_t)QD_ * KD_ * 2);
    _Float16* wvT = (_Float16*)base;    base = alignup(base + (size_t)QD_ * KD_ * 2);
    _Float16* wqT = (_Float16*)base;    base = alignup(base + (size_t)QD_ * QD_ * 2);
    _Float16* woT = (_Float16*)base;    base = alignup(base + (size_t)QD_ * QD_ * 2);
    _Float16* w1T = (_Float16*)base;    base = alignup(base + (size_t)D_ * KD_ * 2);
    _Float16* w2T = (_Float16*)base;    base = alignup(base + (size_t)D_ * D_ * 2);

    k_zero_map<<<(N + 255) / 256, 256, 0, stream>>>(map, N);
    k_scatter_map<<<(U + 255) / 256, 256, 0, stream>>>(unids, map, U);
    k_prep<<<512, 256, 0, stream>>>(Wih, Whh, Wk, Wv, Wq, Wo, W1, W2, wih_h,
                                    whh_h, wkT, wvT, wqT, woT, w1T, w2T);

    const size_t gru_lds = (size_t)(16 * MSG_ + 16 * D_) * 2 +
                           (size_t)(16 * D_ + 2 * 16 * TD3_) * 4;   // ~76 KB
    k_gru<<<(U + 15) / 16, 256, gru_lds, stream>>>(memory, unids, umsg, wih_h,
                                                   whh_h, bih, bhh, mem_upd, U);

    const size_t attn_lds =
        (size_t)ROWS_ * KD_ * 2 +            // kin (aliased later by af16/cat/h1)
        (size_t)ROWS_ * 512 * 2 +            // kv16
        (size_t)16 * QD_ * 2 +               // qin
        (size_t)SB_ * QD_ * 4 +              // qv
        (size_t)SB_ * D_ * 4 +               // nf
        (size_t)2 * SB_ * HEADS_ * KNB_ * 4 +// sc, sw
        (size_t)SB_ * 4 + ROWS_ + 64;        // flags, mask bytes, slack
    k_attn<<<S / SB_, 256, attn_lds, stream>>>(staticf, memory, map, mem_upd,
                                               seeds, stimes, nnids, ntimes,
                                               nfeats, nmask, w_t, b_t, wkT, wvT,
                                               wqT, woT, w1T, w2T, b1, b2,
                                               (float*)d_out);
}
